// BiAttention_dct_12927851561585
// MI455X (gfx1250) — matrix operand order repaired and checked by reading, not compiled, not run on hardware
//
#include <hip/hip_runtime.h>

typedef _Float16 v16h __attribute__((ext_vector_type(16)));
typedef _Float16 h8  __attribute__((ext_vector_type(8)));
typedef float    v8f __attribute__((ext_vector_type(8)));

static constexpr int   DDIM  = 640;
static constexpr int   HID   = 1280;
static constexpr float NEGV  = -65504.0f;
static constexpr float SCALE = 0.03952847075210474f; // 1/sqrt(640)

// ---------------------------------------------------------------------------
// Fragment loaders for v_wmma_f32_16x16x32_f16 (wave32).
// A (16x32, MxK): lane&15 = M row; 16 halfs cover K with the segmented pattern
//   k = h + (h>=8 ? 8 : 0) + (lane>=16 ? 8 : 0)  ->  two contiguous 8-half
//   (16B) LDS loads per lane from a row-major tile.
// B (32xN=16, KxN): lane&15 = N column; 16 halfs cover K with the same
//   segmented pattern as A -> strided gather down column n of row-major B.
// ---------------------------------------------------------------------------
__device__ __forceinline__ v16h load_afrag(const _Float16* base, int rowStride,
                                           int m0, int kbase) {
  int l = threadIdx.x & 31;
  const _Float16* rp = base + (m0 + (l & 15)) * rowStride + kbase + ((l & 16) ? 8 : 0);
  h8 lo = *(const h8*)rp;
  h8 hi = *(const h8*)(rp + 16);
  v16h r;
#pragma unroll
  for (int i = 0; i < 8; ++i) { r[i] = lo[i]; r[i + 8] = hi[i]; }
  return r;
}

__device__ __forceinline__ v16h load_bfrag(const _Float16* base, int rowStride,
                                           int kbase, int nbase) {
  int l = threadIdx.x & 31;
  const _Float16* rp = base + (kbase + ((l & 16) ? 8 : 0)) * rowStride + nbase + (l & 15);
  v16h r;
#pragma unroll
  for (int i = 0; i < 8; ++i) { r[i] = rp[i * rowStride]; r[i + 8] = rp[(i + 16) * rowStride]; }
  return r;
}

__device__ __forceinline__ v8f wmma16(v16h a, v16h b, v8f c) {
  return __builtin_amdgcn_wmma_f32_16x16x32_f16(false, a, false, b, (short)0, c,
                                                false, false);
}

// ---------------------------------------------------------------------------
// f32 -> f16 elementwise convert (weights)
// ---------------------------------------------------------------------------
__global__ __launch_bounds__(256) void f32_to_f16_kernel(
    const float* __restrict__ src, _Float16* __restrict__ dst, int n) {
  int i = blockIdx.x * 256 + threadIdx.x;
  if (i < n) dst[i] = (_Float16)src[i];
}

// ---------------------------------------------------------------------------
// Attention: out = softmax(mask(Q @ K^T / sqrt(D))) @ V  (K == V)
// Block = (batch, 32 query rows), 256 threads = 8 waves.
// ---------------------------------------------------------------------------
template <int NK>
__global__ __launch_bounds__(256) void attn_kernel(
    const float* __restrict__ Qg,            // [B, M, 640]
    const float* __restrict__ Kg,            // [B, NK, 640]
    const unsigned char* __restrict__ maskg, // [B, NK] (bool bytes)
    _Float16* __restrict__ outg,             // [B, M, 640] f16
    int M) {
  constexpr int TPW   = NK / 64;   // S-tiles per wave (2 or 8)
  constexpr int KTROW = NK + 8;    // halfs; Kt chunk row stride (16B aligned)
  constexpr int PROW  = NK + 8;
  constexpr int VROW  = 72;

  extern __shared__ char smem[];
  _Float16* Qc  = (_Float16*)smem;            // 32 x 72
  _Float16* Ktc = Qc + 32 * 72;               // 64 x KTROW (transposed K chunk)
  float*    S   = (float*)(Ktc + 64 * KTROW); // 32 x NK
  _Float16* P   = (_Float16*)(S + 32 * NK);   // 32 x PROW
  _Float16* Vc  = P + 32 * PROW;              // NK x 72 (row-major V chunk)

  const int tid  = threadIdx.x;
  const int w    = tid >> 5;
  const int lane = tid & 31;
  const int chunks = M / 32;
  const int b  = blockIdx.x / chunks;
  const int m0 = (blockIdx.x % chunks) * 32;

  const float* Qrow = Qg + ((size_t)b * M + m0) * DDIM;
  const float* Krow = Kg + (size_t)b * NK * DDIM;
  const unsigned char* mk = maskg + (size_t)b * NK;

  // ---------------- Phase A: S = Q @ K^T ----------------
  const v8f vzero = {};
  v8f acc[TPW];
#pragma unroll
  for (int i = 0; i < TPW; ++i) acc[i] = vzero;

  const int qt = w & 1;
  for (int d0 = 0; d0 < DDIM; d0 += 64) {
    __syncthreads();
    for (int idx = tid; idx < 32 * 64; idx += 256) {
      int r = idx >> 6, c = idx & 63;
      Qc[r * 72 + c] = (_Float16)Qrow[(size_t)r * DDIM + d0 + c];
    }
    for (int idx = tid; idx < 64 * NK; idx += 256) {
      int key = idx >> 6, dl = idx & 63;   // coalesced read along d
      Ktc[dl * KTROW + key] = (_Float16)Krow[(size_t)key * DDIM + d0 + dl];
    }
    __syncthreads();
#pragma unroll
    for (int ds = 0; ds < 2; ++ds) {
      v16h a = load_afrag(Qc, 72, qt * 16, ds * 32);
#pragma unroll
      for (int i = 0; i < TPW; ++i) {
        int kt = (w >> 1) + 4 * i;         // tile t = w + 8i
        v16h bf = load_bfrag(Ktc, KTROW, ds * 32, kt * 16);
        acc[i] = wmma16(a, bf, acc[i]);
      }
    }
  }
  __syncthreads();
#pragma unroll
  for (int i = 0; i < TPW; ++i) {
    int kt = (w >> 1) + 4 * i;
#pragma unroll
    for (int r = 0; r < 8; ++r) {
      int m = r + ((lane & 16) ? 8 : 0);
      int n = lane & 15;
      S[(qt * 16 + m) * NK + kt * 16 + n] = acc[i][r];
    }
  }
  __syncthreads();

  // ---------------- masked softmax rows of S -> P (f16) ----------------
  {
    int row = tid >> 3;   // 32 rows, 8 lanes per row
    int sub = tid & 7;
    float mx = -3.0e38f;
    for (int k = sub; k < NK; k += 8) {
      float v = mk[k] ? NEGV : S[row * NK + k] * SCALE;
      S[row * NK + k] = v;
      mx = fmaxf(mx, v);
    }
#pragma unroll
    for (int off = 4; off > 0; off >>= 1) mx = fmaxf(mx, __shfl_xor(mx, off, 8));
    float sum = 0.0f;
    for (int k = sub; k < NK; k += 8) {
      float e = __expf(S[row * NK + k] - mx);
      sum += e;
      P[row * PROW + k] = (_Float16)e;
    }
#pragma unroll
    for (int off = 4; off > 0; off >>= 1) sum += __shfl_xor(sum, off, 8);
    float rinv = 1.0f / sum;
    for (int k = sub; k < NK; k += 8)
      P[row * PROW + k] = (_Float16)((float)P[row * PROW + k] * rinv);
  }
  __syncthreads();

  // ---------------- Phase B: O = P @ V ----------------
  _Float16* orow = outg + ((size_t)b * M + m0) * DDIM;
  const int nt = w >> 1;
  for (int d0 = 0; d0 < DDIM; d0 += 64) {
    for (int idx = tid; idx < NK * 64; idx += 256) {
      int key = idx >> 6, dl = idx & 63;
      Vc[key * VROW + dl] = (_Float16)Krow[(size_t)key * DDIM + d0 + dl];
    }
    __syncthreads();
    v8f o = vzero;
#pragma unroll
    for (int ks = 0; ks < NK / 32; ++ks) {
      v16h a  = load_afrag(P, PROW, qt * 16, ks * 32);
      v16h bf = load_bfrag(Vc, VROW, ks * 32, nt * 16);
      o = wmma16(a, bf, o);
    }
#pragma unroll
    for (int r = 0; r < 8; ++r) {
      int m = qt * 16 + r + ((lane & 16) ? 8 : 0);
      int d = d0 + nt * 16 + (lane & 15);
      orow[(size_t)m * DDIM + d] = (_Float16)o[r];
    }
    __syncthreads();
  }
}

// ---------------------------------------------------------------------------
// AttFlat logits: logit[b,m] = relu(x[b,m,:]@W1 + b1) @ W2 + b2
// Block = (batch, 64 rows). X tile LDS-resident; hidden streamed 32-wide.
// ---------------------------------------------------------------------------
__global__ __launch_bounds__(256) void attflat_logits_kernel(
    const _Float16* __restrict__ x,    // [B, M, 640] f16
    const _Float16* __restrict__ w1h,  // [640, 1280] f16
    const float* __restrict__ b1,      // [1280]
    const float* __restrict__ w2,      // [1280]
    const float* __restrict__ b2,      // [1]
    float* __restrict__ logits,        // [B, M]
    int M) {
  constexpr int XROW = 648;  // 640 + 8 pad halfs (1296B, 16B aligned)
  constexpr int WROW = 40;   // 32 + 8 pad halfs (80B, 16B aligned)
  extern __shared__ char smem[];
  _Float16* X  = (_Float16*)smem;            // 64 x XROW
  _Float16* Wc = X + 64 * XROW;              // 640 x WROW
  float*    lg = (float*)(Wc + 640 * WROW);  // 64

  const int tid  = threadIdx.x;
  const int w    = tid >> 5;
  const int lane = tid & 31;
  const int chunks = M / 64;
  const int b  = blockIdx.x / chunks;
  const int m0 = (blockIdx.x % chunks) * 64;

  if (tid < 64) lg[tid] = 0.0f;

  { // X tile: 64*640 halfs contiguous in global -> padded LDS rows (uint4)
    const uint4* src = (const uint4*)(x + ((size_t)b * M + m0) * DDIM);
    for (int idx = tid; idx < 64 * 80; idx += 256) {
      int r = idx / 80, c8 = idx % 80;
      *(uint4*)(X + r * XROW + c8 * 8) = src[idx];
    }
  }
  __syncthreads();

  const int rt = w & 3;   // row tile 0..3
  const int jc = w >> 2;  // hidden half-chunk 0..1
  float acc[8];
#pragma unroll
  for (int r = 0; r < 8; ++r) acc[r] = 0.0f;

  for (int ji = 0; ji < HID / 32; ++ji) {
    const int h0 = ji * 32;
    __syncthreads();
    // Wc[d][hh] = W1[d][h0+hh] : B matrix rows (k=d) contiguous in n=hidden
    for (int idx = tid; idx < 640 * 4; idx += 256) {
      int d = idx >> 2, v = idx & 3;
      *(uint4*)(Wc + d * WROW + v * 8) =
          *(const uint4*)(w1h + (size_t)d * HID + h0 + v * 8);
    }
    __syncthreads();
    v8f h = {};
#pragma unroll
    for (int ds = 0; ds < DDIM / 32; ++ds) {
      v16h a  = load_afrag(X, XROW, rt * 16, ds * 32);
      v16h bf = load_bfrag(Wc, WROW, ds * 32, jc * 16);
      h = wmma16(a, bf, h);
    }
    int hid = h0 + jc * 16 + (lane & 15);
    float bias = b1[hid], w2v = w2[hid];
#pragma unroll
    for (int r = 0; r < 8; ++r) {
      float v = h[r] + bias;
      v = v > 0.0f ? v : 0.0f;   // relu
      acc[r] += v * w2v;
    }
  }
#pragma unroll
  for (int r = 0; r < 8; ++r) {
#pragma unroll
    for (int off = 8; off > 0; off >>= 1) acc[r] += __shfl_xor(acc[r], off, 16);
  }
  if ((lane & 15) == 0) {
    int mb = rt * 16 + ((lane & 16) ? 8 : 0);
#pragma unroll
    for (int r = 0; r < 8; ++r) atomicAdd(&lg[mb + r], acc[r]);
  }
  __syncthreads();
  if (tid < 64) logits[(size_t)b * M + m0 + tid] = lg[tid] + b2[0];
}

// ---------------------------------------------------------------------------
// Masked softmax over M rows per batch (axis=1 of att); optional copy output.
// ---------------------------------------------------------------------------
__global__ __launch_bounds__(256) void masked_softmax_kernel(
    const float* __restrict__ logits, const unsigned char* __restrict__ maskg,
    float* __restrict__ att, float* __restrict__ att_copy, int M) {
  __shared__ float red[256];
  const int b = blockIdx.x;
  const int tid = threadIdx.x;
  float v[2];
  int cnt = 0;
  float mx = -3.0e38f;
  for (int i = tid; i < M; i += 256) {
    float t = maskg[(size_t)b * M + i] ? NEGV : logits[(size_t)b * M + i];
    v[cnt++] = t;
    mx = fmaxf(mx, t);
  }
  red[tid] = mx; __syncthreads();
  for (int s = 128; s > 0; s >>= 1) {
    if (tid < s) red[tid] = fmaxf(red[tid], red[tid + s]);
    __syncthreads();
  }
  mx = red[0]; __syncthreads();
  float sum = 0.0f;
  for (int c = 0; c < cnt; ++c) { v[c] = __expf(v[c] - mx); sum += v[c]; }
  red[tid] = sum; __syncthreads();
  for (int s = 128; s > 0; s >>= 1) {
    if (tid < s) red[tid] += red[tid + s];
    __syncthreads();
  }
  float rinv = 1.0f / red[0];
  cnt = 0;
  for (int i = tid; i < M; i += 256) {
    float a = v[cnt++] * rinv;
    att[(size_t)b * M + i] = a;
    if (att_copy) att_copy[(size_t)b * M + i] = a;
  }
}

// ---------------------------------------------------------------------------
// pooled[b,d] = sum_m att[b,m] * x[b,m,d]
// ---------------------------------------------------------------------------
__global__ __launch_bounds__(256) void pooled_kernel(
    const float* __restrict__ att, const _Float16* __restrict__ x,
    float* __restrict__ pooled, int M) {
  __shared__ float sa[512];
  const int b = blockIdx.x;
  const int tid = threadIdx.x;
  for (int i = tid; i < M; i += 256) sa[i] = att[(size_t)b * M + i];
  __syncthreads();
  float a0 = 0.0f, a1 = 0.0f, a2 = 0.0f;
  const _Float16* xb = x + (size_t)b * M * DDIM;
  for (int m = 0; m < M; ++m) {
    float av = sa[m];
    const _Float16* xr = xb + (size_t)m * DDIM;
    a0 += av * (float)xr[tid];
    a1 += av * (float)xr[tid + 256];
    if (tid < 128) a2 += av * (float)xr[tid + 512];
  }
  pooled[(size_t)b * DDIM + tid] = a0;
  pooled[(size_t)b * DDIM + tid + 256] = a1;
  if (tid < 128) pooled[(size_t)b * DDIM + tid + 512] = a2;
}

// ---------------------------------------------------------------------------
// out[b,j] = bm[j] + sum_d pooled[b,d] * Wm[d,j]   (tiny 64x640x640 GEMM)
// ---------------------------------------------------------------------------
__global__ __launch_bounds__(256) void final_linear_kernel(
    const float* __restrict__ pooled, const float* __restrict__ Wm,
    const float* __restrict__ bm, float* __restrict__ out) {
  __shared__ float pl[DDIM];
  const int b = blockIdx.x;
  const int tid = threadIdx.x;
  for (int i = tid; i < DDIM; i += 256) pl[i] = pooled[(size_t)b * DDIM + i];
  __syncthreads();
  for (int j = tid; j < DDIM; j += 256) {
    float accv = bm[j];
    for (int d = 0; d < DDIM; ++d) accv += pl[d] * Wm[(size_t)d * DDIM + j];
    out[(size_t)b * DDIM + j] = accv;
  }
}

// ---------------------------------------------------------------------------
extern "C" void kernel_launch(void* const* d_in, const int* in_sizes, int n_in,
                              void* d_out, int out_size, void* d_ws, size_t ws_size,
                              hipStream_t stream) {
  (void)in_sizes; (void)n_in; (void)out_size; (void)ws_size;
  const float* i_batch = (const float*)d_in[0];
  const float* q_batch = (const float*)d_in[1];
  const unsigned char* i_mask = (const unsigned char*)d_in[2];
  const unsigned char* q_mask = (const unsigned char*)d_in[3];
  const float* lW1 = (const float*)d_in[4];
  const float* lb1 = (const float*)d_in[5];
  const float* lW2 = (const float*)d_in[6];
  const float* lb2 = (const float*)d_in[7];
  const float* lWm = (const float*)d_in[8];
  const float* lbm = (const float*)d_in[9];
  const float* iW1 = (const float*)d_in[10];
  const float* ib1 = (const float*)d_in[11];
  const float* iW2 = (const float*)d_in[12];
  const float* ib2 = (const float*)d_in[13];
  const float* iWm = (const float*)d_in[14];
  const float* ibm = (const float*)d_in[15];

  char* ws = (char*)d_ws;
  size_t off = 0;
  auto carve = [&](size_t bytes) {
    void* p = ws + off;
    off += (bytes + 255) & ~(size_t)255;
    return p;
  };
  _Float16* xi   = (_Float16*)carve(64UL * 512 * 640 * 2); // attn1 out
  _Float16* xl   = (_Float16*)carve(64UL * 128 * 640 * 2); // attn2 out
  _Float16* w1hl = (_Float16*)carve(640UL * 1280 * 2);
  _Float16* w1hi = (_Float16*)carve(640UL * 1280 * 2);
  float* lg_i = (float*)carve(64UL * 512 * 4);
  float* lg_l = (float*)carve(64UL * 128 * 4);
  float* at_i = (float*)carve(64UL * 512 * 4);
  float* at_l = (float*)carve(64UL * 128 * 4);
  float* pl_i = (float*)carve(64UL * 640 * 4);
  float* pl_l = (float*)carve(64UL * 640 * 4);

  float* out_i_feat = (float*)d_out;
  float* out_l_feat = out_i_feat + 64 * 640;
  float* out_i_w    = out_l_feat + 64 * 640;

  const size_t sm_attn1 = (size_t)32 * 72 * 2 + (size_t)64 * (128 + 8) * 2 +
                          (size_t)32 * 128 * 4 + (size_t)32 * (128 + 8) * 2 +
                          (size_t)128 * 72 * 2;                       // 65,536 B
  const size_t sm_attn2 = (size_t)32 * 72 * 2 + (size_t)64 * (512 + 8) * 2 +
                          (size_t)32 * 512 * 4 + (size_t)32 * (512 + 8) * 2 +
                          (size_t)512 * 72 * 2;                       // 243,712 B
  const size_t sm_flat = (size_t)64 * 648 * 2 + (size_t)640 * 40 * 2 + 64 * 4;

  {
    auto* f1 = attn_kernel<128>;
    auto* f2 = attn_kernel<512>;
    hipFuncSetAttribute((const void*)f1, hipFuncAttributeMaxDynamicSharedMemorySize,
                        (int)sm_attn1);
    hipFuncSetAttribute((const void*)f2, hipFuncAttributeMaxDynamicSharedMemorySize,
                        (int)sm_attn2);
    hipFuncSetAttribute((const void*)attflat_logits_kernel,
                        hipFuncAttributeMaxDynamicSharedMemorySize, (int)sm_flat);
  }

  f32_to_f16_kernel<<<(640 * 1280 + 255) / 256, 256, 0, stream>>>(lW1, w1hl, 640 * 1280);
  f32_to_f16_kernel<<<(640 * 1280 + 255) / 256, 256, 0, stream>>>(iW1, w1hi, 640 * 1280);

  // attention 1: Q=i_batch (512 rows), K=V=q_batch (128 keys), mask over keys
  attn_kernel<128><<<64 * (512 / 32), 256, sm_attn1, stream>>>(
      i_batch, q_batch, q_mask, xi, 512);
  // attention 2: Q=q_batch (128 rows), K=V=i_batch (512 keys)
  attn_kernel<512><<<64 * (128 / 32), 256, sm_attn2, stream>>>(
      q_batch, i_batch, i_mask, xl, 128);

  attflat_logits_kernel<<<64 * (512 / 64), 256, sm_flat, stream>>>(
      xi, w1hl, lb1, lW2, lb2, lg_i, 512);
  attflat_logits_kernel<<<64 * (128 / 64), 256, sm_flat, stream>>>(
      xl, w1hi, ib1, iW2, ib2, lg_l, 128);

  masked_softmax_kernel<<<64, 256, 0, stream>>>(lg_i, i_mask, at_i, out_i_w, 512);
  masked_softmax_kernel<<<64, 256, 0, stream>>>(lg_l, q_mask, at_l, nullptr, 128);

  pooled_kernel<<<64, 256, 0, stream>>>(at_i, xi, pl_i, 512);
  pooled_kernel<<<64, 256, 0, stream>>>(at_l, xl, pl_l, 128);

  final_linear_kernel<<<64, 256, 0, stream>>>(pl_i, lWm, lbm, out_i_feat);
  final_linear_kernel<<<64, 256, 0, stream>>>(pl_l, iWm, ibm, out_l_feat);
}
